// Model_11879879543293
// MI455X (gfx1250) — matrix operand order repaired and checked by reading, not compiled, not run on hardware
//
#include <hip/hip_runtime.h>

typedef __attribute__((ext_vector_type(16))) _Float16 v16h;
typedef __attribute__((ext_vector_type(8)))  _Float16 h8;
typedef __attribute__((ext_vector_type(4)))  _Float16 h4;
typedef __attribute__((ext_vector_type(8)))  float    v8f;
typedef __attribute__((ext_vector_type(4)))  float    f4;

namespace {
constexpr int kB = 4, kH = 16, kS = 2048, kD = 128;
constexpr int QTILE   = 128;  // query rows per block (8 waves x 16)
constexpr int KTILE   = 32;   // keys per LDS tile
constexpr int KSTRIDE = 136;  // halfs per K-tile row (128 + 8 pad -> bank rotate)
constexpr int VSTRIDE = 40;   // halfs per Vt row (32 + 8 pad -> bank rotate)
constexpr float SCALE = 0.08838834764831845f; // 1/sqrt(128)
}

static __device__ __forceinline__ v16h cat8(h8 lo, h8 hi) {
  return __builtin_shufflevector(lo, hi, 0,1,2,3,4,5,6,7,8,9,10,11,12,13,14,15);
}

__global__ __launch_bounds__(256) void
fa_fwd(const float* __restrict__ Q, const float* __restrict__ K,
       const float* __restrict__ V, const float* __restrict__ M,
       float* __restrict__ O)
{
  __shared__ _Float16 Ksh[KTILE * KSTRIDE];  // K tile row-major [key][d]
  __shared__ _Float16 Vt [kD    * VSTRIDE];  // V tile transposed [d][key]

  const int tid  = threadIdx.x;
  const int lane = tid & 31;
  const int wave = tid >> 5;
  const int ln16 = lane & 15;
  const int hf   = lane >> 4;           // which 16-lane half

  const int bh = blockIdx.y;            // b*H + h
  const int q0 = blockIdx.x * QTILE + wave * 16;
  const int q  = q0 + ln16;             // this lane's query row

  const size_t bh_qkv = (size_t)bh * kS * kD;
  const size_t mrow   = ((size_t)bh * kS + q) * kS; // mask row base for this lane

  // Q fragments in WMMA-B layout: frag c, element e -> d = c*32 + hf*16 + e
  v16h qf[4];
  {
    const float* qp = Q + bh_qkv + (size_t)q * kD;
#pragma unroll
    for (int c = 0; c < 4; ++c)
#pragma unroll
      for (int e = 0; e < 16; ++e)
        qf[c][e] = (_Float16)qp[c * 32 + hf * 16 + e];
  }

  v8f oacc[8];                          // O^T accumulators: elem r -> d = nb*16 + r + 8*hf
#pragma unroll
  for (int i = 0; i < 8; ++i) oacc[i] = v8f{};
  float rmax = -3.0e38f, rsum = 0.0f;

  const int nkt = kS / KTILE;
  for (int kt = 0; kt < nkt; ++kt) {
    __syncthreads();
    // Cooperative tile load: K -> f16 row-major, V -> f16 transposed
    {
      int idx = tid;                    // 4 iters x 256 thr = 1024 float4 = 32x128 f32
#pragma unroll
      for (int i = 0; i < 4; ++i, idx += 256) {
        const int row = idx >> 5;               // key within tile
        const int col = (idx & 31) << 2;        // d
        const size_t g = bh_qkv + (size_t)(kt * KTILE + row) * kD + col;
        f4 kv = *(const f4*)(K + g);
        f4 vv = *(const f4*)(V + g);
        h4 kh;
        kh.x = (_Float16)kv.x; kh.y = (_Float16)kv.y;
        kh.z = (_Float16)kv.z; kh.w = (_Float16)kv.w;
        *(h4*)(&Ksh[row * KSTRIDE + col]) = kh;
        Vt[(col + 0) * VSTRIDE + row] = (_Float16)vv.x;
        Vt[(col + 1) * VSTRIDE + row] = (_Float16)vv.y;
        Vt[(col + 2) * VSTRIDE + row] = (_Float16)vv.z;
        Vt[(col + 3) * VSTRIDE + row] = (_Float16)vv.w;
      }
      if (kt + 1 < nkt) {               // prefetch next tile: 64B/thread covers 16KB
        const size_t gn = bh_qkv + (size_t)((kt + 1) * KTILE) * kD + tid * 16;
        __builtin_prefetch(K + gn, 0, 3);
        __builtin_prefetch(V + gn, 0, 3);
      }
    }
    __syncthreads();

    // S^T = K * Q^T, two 16-key blocks. C layout: lane->q, elem r -> key blk*16 + r + 8*hf
    v8f st0 = v8f{}, st1 = v8f{};
#pragma unroll
    for (int c = 0; c < 4; ++c) {
      const _Float16* k0 = &Ksh[ln16 * KSTRIDE + c * 32 + hf * 16];
      v16h a0 = cat8(*(const h8*)k0, *(const h8*)(k0 + 8));
      const _Float16* k1 = &Ksh[(16 + ln16) * KSTRIDE + c * 32 + hf * 16];
      v16h a1 = cat8(*(const h8*)k1, *(const h8*)(k1 + 8));
      st0 = __builtin_amdgcn_wmma_f32_16x16x32_f16(false, a0, false, qf[c],
                                                   (short)0, st0, false, false);
      st1 = __builtin_amdgcn_wmma_f32_16x16x32_f16(false, a1, false, qf[c],
                                                   (short)0, st1, false, false);
    }

    // Scale + additive mask (keys contiguous along element index)
    const int kbase = kt * KTILE + 8 * hf;
#pragma unroll
    for (int r = 0; r < 8; ++r) {
      st0[r] = st0[r] * SCALE + M[mrow + kbase + r];
      st1[r] = st1[r] * SCALE + M[mrow + kbase + 16 + r];
    }

    // Online softmax: stats are per-lane (q on lane); halves hold disjoint key subsets
    float mx = st0[0];
#pragma unroll
    for (int r = 1; r < 8; ++r) mx = fmaxf(mx, st0[r]);
#pragma unroll
    for (int r = 0; r < 8; ++r) mx = fmaxf(mx, st1[r]);
    mx = fmaxf(mx, __shfl_xor(mx, 16, 32));
    const float mnew = fmaxf(rmax, mx);
    const float corr = __expf(rmax - mnew);
    rmax = mnew;

    float p0[8], p1[8], ts = 0.0f;
#pragma unroll
    for (int r = 0; r < 8; ++r) { p0[r] = __expf(st0[r] - mnew); ts += p0[r]; }
#pragma unroll
    for (int r = 0; r < 8; ++r) { p1[r] = __expf(st1[r] - mnew); ts += p1[r]; }
    ts += __shfl_xor(ts, 16, 32);
    rsum = rsum * corr + ts;

    // Build P^T B-fragment: lane->q, element e -> key = hf*16 + e
    v16h pb;
#pragma unroll
    for (int r = 0; r < 8; ++r) {
      const float o0 = __shfl_xor(p0[r], 16, 32);  // other half's st0 prob (row r)
      const float o1 = __shfl_xor(p1[r], 16, 32);  // other half's st1 prob (row r)
      pb[r]     = (_Float16)(hf ? o1 : p0[r]);     // keys hf*16 + 0..7
      pb[r + 8] = (_Float16)(hf ? p1[r] : o0);     // keys hf*16 + 8..15
    }

    // O^T = corr*O^T + V^T * P^T ; A = V^T chunk (16 d-rows x 32 keys)
#pragma unroll
    for (int nb = 0; nb < 8; ++nb) {
      const _Float16* vp = &Vt[(nb * 16 + ln16) * VSTRIDE + hf * 16];
      v16h va = cat8(*(const h8*)vp, *(const h8*)(vp + 8));
      v8f acc = oacc[nb];
#pragma unroll
      for (int r = 0; r < 8; ++r) acc[r] *= corr;
      oacc[nb] = __builtin_amdgcn_wmma_f32_16x16x32_f16(false, va, false, pb,
                                                        (short)0, acc, false, false);
    }
  }

  // Epilogue: normalize and store; elem r -> d = nb*16 + 8*hf + r (contiguous 8)
  const float inv = 1.0f / rsum;
  float* op = O + bh_qkv + (size_t)q * kD;
#pragma unroll
  for (int nb = 0; nb < 8; ++nb)
#pragma unroll
    for (int r = 0; r < 8; ++r)
      op[nb * 16 + hf * 8 + r] = oacc[nb][r] * inv;
}

extern "C" void kernel_launch(void* const* d_in, const int* in_sizes, int n_in,
                              void* d_out, int out_size, void* d_ws, size_t ws_size,
                              hipStream_t stream) {
  const float* Q = (const float*)d_in[0];
  const float* K = (const float*)d_in[1];
  const float* V = (const float*)d_in[2];
  const float* M = (const float*)d_in[3];
  float* O = (float*)d_out;
  (void)in_sizes; (void)n_in; (void)out_size; (void)d_ws; (void)ws_size;
  dim3 grid(kS / QTILE, kB * kH);
  fa_fwd<<<grid, dim3(256), 0, stream>>>(Q, K, V, M, O);
}
